// BaseDecoder_45518063403269
// MI455X (gfx1250) — matrix operand order repaired and checked by reading, not compiled, not run on hardware
//
#include <hip/hip_runtime.h>
#include <hip/hip_bf16.h>

typedef __attribute__((ext_vector_type(16))) _Float16 v16h;
typedef __attribute__((ext_vector_type(8)))  float    v8f;

#define NLAYER 6
#define EDIM   1024
#define HEADS  16
#define DH     64
#define FFDIM  4096
#define BATCH  16
#define LQ     128
#define LS     128
#define V0C    80
#define OUTCOL 208          // V0 + L
#define NTOK   2048         // B*L
#define NEGINF (-__builtin_inff())

// ---------------------------------------------------------------------------
// Generic batched WMMA GEMM:  C = act( scale * A @ B^T + bias )
//   A[m,k] = Abase[row*rsA + k]            (k contiguous)
//   B[n,k] = Bbase[n*rsB + k*csB]          (csB==1 -> CONTIG_B specialization)
//   batch z -> (b = z/nh, h = z%nh); per-operand (b,h) strides.
// Block = 128 threads (4 waves); block tile = 16 rows x 64 cols; each wave
// owns one 16x16 tile. The 16x64 A tile is staged once per block into LDS
// via GLOBAL_LOAD_ASYNC_TO_LDS_B128 (ASYNCcnt), double buffered; K advances
// 64 per iteration = 2 WMMAs per wave.
// ---------------------------------------------------------------------------
template <bool CONTIG_B>
__global__ __launch_bounds__(128)
void gemm_wmma_k(const float* __restrict__ A, long long sAb, long long sAh, int rsA,
                 const float* __restrict__ Bm, long long sBb, long long sBh, int rsB, int csB,
                 const float* __restrict__ bias,
                 float* __restrict__ C, long long sCb, long long sCh, int ldc,
                 int M, int N, int K, int nh, float scale, int act)
{
    __shared__ float Asf[2][16 * 64];          // 2 x 4 KB, double buffered (raw f32)

    const int z = blockIdx.z;
    const int b = z / nh, h = z - b * nh;
    const float* Ab = A  + (long long)b * sAb + (long long)h * sAh;
    const float* Bb = Bm + (long long)b * sBb + (long long)h * sBh;
    float*       Cb = C  + (long long)b * sCb + (long long)h * sCh;

    const int tid  = threadIdx.x;
    const int wave = tid >> 5;
    const int lane = tid & 31;
    const int row0 = blockIdx.y * 16;
    const int col0 = blockIdx.x * 64 + wave * 16;
    const bool live = (col0 < N);   // wave-uniform; dead waves still help stage A

    // WMMA fragment coordinates (A/B k order is contiguous: K = 16*hi + i)
    const int m   = lane & 15;      // A row within tile / C column lane
    const int hi  = lane >> 4;      // lane half
    const int kbA = hi * 16;        // A: elem i -> K = kbA + i
    const int kbB = hi * 16;        // B: elem i -> K = kbB + i
    const int nn  = lane & 15;

    // cooperative A staging: thread t owns row (t>>3), 8 floats at col (t&7)*8
    const int rA = tid >> 3;
    const int cA = (tid & 7) * 8;
    const float* asrc = Ab + (long long)(row0 + rA) * rsA + cA;
    const float* brow = Bb + (long long)(col0 + nn) * rsB;

    v8f acc = {};
    int buf = 0;
    for (int k0 = 0; k0 < K; k0 += 64, buf ^= 1) {
        // ---- async-copy A tile slice (32 bytes/thread) into LDS ----
        {
            unsigned           dst = (unsigned)(unsigned long long)(&Asf[buf][rA * 64 + cA]);
            unsigned long long src = (unsigned long long)(asrc + k0);
            // INST_OFFSET is added to both the LDS and the memory address
            asm volatile("global_load_async_to_lds_b128 %0, %1, off"
                         :: "v"(dst), "v"(src) : "memory");
            asm volatile("global_load_async_to_lds_b128 %0, %1, off offset:16"
                         :: "v"(dst), "v"(src) : "memory");
            asm volatile("s_wait_asynccnt 0x0" ::: "memory");
        }
        __syncthreads();

        if (live) {
            #pragma unroll
            for (int kk = 0; kk < 64; kk += 32) {
                // A fragment from LDS (f32), convert to f16 in-register
                const float* ap = &Asf[buf][m * 64 + kk + kbA];
                float4 a0 = ((const float4*)ap)[0];
                float4 a1 = ((const float4*)ap)[1];
                float4 a2 = ((const float4*)(ap + 8))[0];
                float4 a3 = ((const float4*)(ap + 8))[1];
                v16h av;
                av[0]=(_Float16)a0.x;  av[1]=(_Float16)a0.y;  av[2]=(_Float16)a0.z;  av[3]=(_Float16)a0.w;
                av[4]=(_Float16)a1.x;  av[5]=(_Float16)a1.y;  av[6]=(_Float16)a1.z;  av[7]=(_Float16)a1.w;
                av[8]=(_Float16)a2.x;  av[9]=(_Float16)a2.y;  av[10]=(_Float16)a2.z; av[11]=(_Float16)a2.w;
                av[12]=(_Float16)a3.x; av[13]=(_Float16)a3.y; av[14]=(_Float16)a3.z; av[15]=(_Float16)a3.w;

                // B fragment from global (convert f32 -> f16 in-register)
                v16h bv;
                if (CONTIG_B) {
                    const float4* p = (const float4*)(brow + k0 + kk + kbB);
                    float4 b0 = p[0], b1 = p[1], b2 = p[2], b3 = p[3];
                    bv[0]=(_Float16)b0.x;  bv[1]=(_Float16)b0.y;  bv[2]=(_Float16)b0.z;  bv[3]=(_Float16)b0.w;
                    bv[4]=(_Float16)b1.x;  bv[5]=(_Float16)b1.y;  bv[6]=(_Float16)b1.z;  bv[7]=(_Float16)b1.w;
                    bv[8]=(_Float16)b2.x;  bv[9]=(_Float16)b2.y;  bv[10]=(_Float16)b2.z; bv[11]=(_Float16)b2.w;
                    bv[12]=(_Float16)b3.x; bv[13]=(_Float16)b3.y; bv[14]=(_Float16)b3.z; bv[15]=(_Float16)b3.w;
                } else {
                    const float* p = brow + (long long)(k0 + kk + kbB) * csB;
                    #pragma unroll
                    for (int i = 0; i < 16; ++i) bv[i] = (_Float16)p[(long long)i * csB];
                }

                acc = __builtin_amdgcn_wmma_f32_16x16x32_f16(false, av, false, bv,
                                                             (short)0, acc, false, false);
            }
        }
    }

    if (live) {
        // C/D layout: VGPR r -> (M = r + 8*hi, N = lane&15)
        #pragma unroll
        for (int r = 0; r < 8; ++r) {
            const int row = row0 + r + hi * 8;
            const int col = col0 + nn;
            float v = acc[r] * scale;
            if (bias) v += bias[col];
            if (act == 1) v = 0.5f * v * (1.0f + erff(v * 0.70710678118654752f)); // exact gelu
            Cb[(long long)row * ldc + col] = v;
        }
    }
}

// ---------------------------------------------------------------------------
// x = tok_emb[seq]*sqrt(E) + branch_emb[branch]*sqrt(E)
// ---------------------------------------------------------------------------
__global__ __launch_bounds__(256)
void embed_kernel(const int* __restrict__ seq, const int* __restrict__ br,
                  const float* __restrict__ tok, const float* __restrict__ bemb,
                  float* __restrict__ x)
{
    long long i = (long long)blockIdx.x * 256 + threadIdx.x;   // NTOK*E
    int e = (int)(i % EDIM);
    long long t = i / EDIM;
    x[i] = 32.0f * (tok[(long long)seq[t] * EDIM + e] + bemb[(long long)br[t] * EDIM + e]);
}

// ---------------------------------------------------------------------------
// bias[b,h,q,k] = dist+up+down+right (or -inf above diagonal)
// ---------------------------------------------------------------------------
__global__ __launch_bounds__(256)
void build_bias(const int* __restrict__ dist, const int* __restrict__ up,
                const int* __restrict__ down, const int* __restrict__ right,
                const float* __restrict__ de, const float* __restrict__ ue,
                const float* __restrict__ we, const float* __restrict__ re,
                float* __restrict__ bias)
{
    long long i = (long long)blockIdx.x * 256 + threadIdx.x;   // B*L*L*H
    int hh = (int)(i % HEADS);
    long long r = i / HEADS;
    int k = (int)(r % LQ); r /= LQ;
    int q = (int)(r % LQ);
    int b = (int)(r / LQ);
    long long sq = (((long long)b * LQ) + q) * LQ + k;
    float v;
    if (k > q) {
        v = NEGINF;
    } else {
        v = de[(long long)dist[sq]  * HEADS + hh] + ue[(long long)up[sq]   * HEADS + hh]
          + we[(long long)down[sq]  * HEADS + hh] + re[(long long)right[sq]* HEADS + hh];
    }
    bias[((((long long)b * HEADS + hh) * LQ) + q) * LQ + k] = v;
}

// memory [S,B,E] -> [B,S,E]
__global__ __launch_bounds__(256)
void transpose_mem(const float* __restrict__ mem, float* __restrict__ out)
{
    long long i = (long long)blockIdx.x * 256 + threadIdx.x;   // S*B*E
    int e = (int)(i % EDIM);
    long long r = i / EDIM;
    int b = (int)(r % BATCH);
    int s = (int)(r / BATCH);
    out[(((long long)b * LS) + s) * EDIM + e] = mem[i];
}

// ---------------------------------------------------------------------------
// Row softmax over 128 keys; optional additive bias [B,H,L,L]; optional
// key-padding from int sequences (==0) or byte mask.
// blockIdx.x = (b*H + h)*L + q ; blockDim = 128
// ---------------------------------------------------------------------------
__global__ __launch_bounds__(128)
void softmax_kernel(float* __restrict__ sc, const float* __restrict__ bias,
                    const int* __restrict__ seq, const unsigned char* __restrict__ kpm)
{
    const int q = blockIdx.x % LQ;
    const int z = blockIdx.x / LQ;
    const int b = z / HEADS;
    const int k = threadIdx.x;
    const long long off = (((long long)z * LQ) + q) * LQ + k;

    float s = sc[off];
    if (bias) s += bias[off];
    if (seq && seq[b * LQ + k] == 0) s = NEGINF;
    if (kpm && kpm[b * LS + k])      s = NEGINF;

    __shared__ float red[128];
    red[k] = s; __syncthreads();
    for (int st = 64; st > 0; st >>= 1) { if (k < st) red[k] = fmaxf(red[k], red[k + st]); __syncthreads(); }
    const float mx = red[0]; __syncthreads();

    float e = (mx == NEGINF) ? 0.0f : __expf(s - mx);
    red[k] = e; __syncthreads();
    for (int st = 64; st > 0; st >>= 1) { if (k < st) red[k] += red[k + st]; __syncthreads(); }
    const float sum = red[0];

    sc[off] = (sum > 0.0f) ? e / sum : 0.0f;
}

// ---------------------------------------------------------------------------
// out = LayerNorm(x + t) over E=1024 (t may be null). One block per row.
// ---------------------------------------------------------------------------
__global__ __launch_bounds__(256)
void add_layernorm(const float* __restrict__ x, const float* __restrict__ t,
                   const float* __restrict__ w, const float* __restrict__ bb,
                   float* __restrict__ out)
{
    const int row = blockIdx.x;
    const int tid = threadIdx.x;
    __shared__ float red[256];

    float v[4];
    #pragma unroll
    for (int i = 0; i < 4; ++i) {
        const int e = tid + i * 256;
        float a = x[(long long)row * EDIM + e];
        if (t) a += t[(long long)row * EDIM + e];
        v[i] = a;
    }
    float s = v[0] + v[1] + v[2] + v[3];
    red[tid] = s; __syncthreads();
    for (int st = 128; st > 0; st >>= 1) { if (tid < st) red[tid] += red[tid + st]; __syncthreads(); }
    const float mean = red[0] * (1.0f / EDIM); __syncthreads();

    float sq = 0.0f;
    #pragma unroll
    for (int i = 0; i < 4; ++i) { const float d = v[i] - mean; sq += d * d; }
    red[tid] = sq; __syncthreads();
    for (int st = 128; st > 0; st >>= 1) { if (tid < st) red[tid] += red[tid + st]; __syncthreads(); }
    const float rinv = rsqrtf(red[0] * (1.0f / EDIM) + 1e-5f); __syncthreads();

    #pragma unroll
    for (int i = 0; i < 4; ++i) {
        const int e = tid + i * 256;
        out[(long long)row * EDIM + e] = (v[i] - mean) * rinv * w[e] + bb[e];
    }
}

// out[i] = pred_mask[i] ? -inf : out[i]
__global__ __launch_bounds__(256)
void mask_kernel(const unsigned char* __restrict__ pm, float* __restrict__ out)
{
    long long i = (long long)blockIdx.x * 256 + threadIdx.x;   // NTOK*OUTCOL
    if (pm[i]) out[i] = NEGINF;
}

// ---------------------------------------------------------------------------
// Host-side launcher
// ---------------------------------------------------------------------------
static void gemm(hipStream_t st,
                 const float* A, long long sAb, long long sAh, int rsA,
                 const float* B, long long sBb, long long sBh, int rsB, int csB,
                 const float* bias,
                 float* C, long long sCb, long long sCh, int ldc,
                 int M, int N, int K, int Z, int nh, float scale, int act)
{
    dim3 grid((N + 63) / 64, M / 16, Z);
    if (csB == 1) {
        gemm_wmma_k<true><<<grid, 128, 0, st>>>(A, sAb, sAh, rsA, B, sBb, sBh, rsB, csB,
                                                bias, C, sCb, sCh, ldc, M, N, K, nh, scale, act);
    } else {
        gemm_wmma_k<false><<<grid, 128, 0, st>>>(A, sAb, sAh, rsA, B, sBb, sBh, rsB, csB,
                                                 bias, C, sCb, sCh, ldc, M, N, K, nh, scale, act);
    }
}

extern "C" void kernel_launch(void* const* d_in, const int* in_sizes, int n_in,
                              void* d_out, int out_size, void* d_ws, size_t ws_size,
                              hipStream_t stream)
{
    const int*   seq   = (const int*)d_in[0];
    const int*   brs   = (const int*)d_in[1];
    const int*   dist  = (const int*)d_in[2];
    const int*   uploc = (const int*)d_in[3];
    const int*   dnloc = (const int*)d_in[4];
    const int*   rtloc = (const int*)d_in[5];
    const unsigned char* pmask = (const unsigned char*)d_in[6];
    const float* memory = (const float*)d_in[7];
    const unsigned char* mkpm = (const unsigned char*)d_in[8];
    const float* tok_emb = (const float*)d_in[9];
    const float* br_emb  = (const float*)d_in[10];
    const float* dist_e  = (const float*)d_in[11];
    const float* up_e    = (const float*)d_in[12];
    const float* dn_e    = (const float*)d_in[13];
    const float* rt_e    = (const float*)d_in[14];
    const float* siw = (const float*)d_in[15];
    const float* sib = (const float*)d_in[16];
    const float* sow = (const float*)d_in[17];
    const float* sob = (const float*)d_in[18];
    const float* ciw = (const float*)d_in[19];
    const float* cib = (const float*)d_in[20];
    const float* cow = (const float*)d_in[21];
    const float* cob = (const float*)d_in[22];
    const float* l1w = (const float*)d_in[23];
    const float* l1b = (const float*)d_in[24];
    const float* l2w = (const float*)d_in[25];
    const float* l2b = (const float*)d_in[26];
    const float* ln1w = (const float*)d_in[27];
    const float* ln1b = (const float*)d_in[28];
    const float* ln2w = (const float*)d_in[29];
    const float* ln2b = (const float*)d_in[30];
    const float* ln3w = (const float*)d_in[31];
    const float* ln3b = (const float*)d_in[32];
    const float* flnw = (const float*)d_in[33];
    const float* flnb = (const float*)d_in[34];
    const float* genw = (const float*)d_in[35];
    const float* genb = (const float*)d_in[36];
    const float* e0w = (const float*)d_in[37];
    const float* e0b = (const float*)d_in[38];
    const float* e1w = (const float*)d_in[39];
    const float* e1b = (const float*)d_in[40];
    float* out = (float*)d_out;

    // ---- workspace layout (floats), ~136 MB total ----
    const long long MF = 1ll << 20;
    float* ws   = (float*)d_ws;
    float* X    = ws;              // [2048,1024]           2M
    float* TMP  = ws + 2 * MF;     // [2048,1024]           2M
    float* QKV  = ws + 4 * MF;     // [2048,3072]           6M  (also E0/E1 at the end)
    float* KVK  = ws + 10 * MF;    // cross K [B,S,E]       2M
    float* KVV  = ws + 12 * MF;    // cross V [B,S,E]       2M
    float* BIAS = ws + 14 * MF;    // [B,H,L,L]             4M
    float* SC   = ws + 18 * MF;    // [B*H,L,L]             4M
    float* FF1  = ws + 22 * MF;    // [2048,4096]           8M
    float* MEMB = ws + 30 * MF;    // [B,S,E]               2M
    float* CTX  = ws + 32 * MF;    // [2048,1024]           2M

    const long long E2 = (long long)EDIM * EDIM;
    const long long sBH = (long long)HEADS * LQ * LQ;   // per-b stride of [B,H,L,L]
    const long long sHH = (long long)LQ * LQ;

    // ---- prologue ----
    embed_kernel<<<(NTOK * EDIM) / 256, 256, 0, stream>>>(seq, brs, tok_emb, br_emb, X);
    build_bias<<<((long long)BATCH * LQ * LQ * HEADS) / 256, 256, 0, stream>>>(
        dist, uploc, dnloc, rtloc, dist_e, up_e, dn_e, rt_e, BIAS);
    transpose_mem<<<(LS * BATCH * EDIM) / 256, 256, 0, stream>>>(memory, MEMB);

    for (int l = 0; l < NLAYER; ++l) {
        const float* Wsi = siw + (long long)l * 3 * E2;
        const float* bsi = sib + (long long)l * 3 * EDIM;
        const float* Wso = sow + (long long)l * E2;
        const float* bso = sob + (long long)l * EDIM;
        const float* Wci = ciw + (long long)l * 3 * E2;
        const float* bci = cib + (long long)l * 3 * EDIM;
        const float* Wco = cow + (long long)l * E2;
        const float* bco = cob + (long long)l * EDIM;

        // ---- self attention ----
        gemm(stream, X, 0, 0, EDIM, Wsi, 0, 0, EDIM, 1, bsi,
             QKV, 0, 0, 3 * EDIM, NTOK, 3 * EDIM, EDIM, 1, 1, 1.0f, 0);
        // scores = Q K^T / 8      (z = b*H + h)
        gemm(stream, QKV, (long long)LQ * 3 * EDIM, DH, 3 * EDIM,
             QKV + EDIM, (long long)LQ * 3 * EDIM, DH, 3 * EDIM, 1, nullptr,
             SC, sBH, sHH, LQ, LQ, LQ, DH, BATCH * HEADS, HEADS, 0.125f, 0);
        softmax_kernel<<<BATCH * HEADS * LQ, 128, 0, stream>>>(SC, BIAS, seq, nullptr);
        // ctx = attn @ V  -> CTX [B,L,H,D]
        gemm(stream, SC, sBH, sHH, LQ,
             QKV + 2 * EDIM, (long long)LQ * 3 * EDIM, DH, 1, 3 * EDIM, nullptr,
             CTX, (long long)LQ * EDIM, DH, EDIM, LQ, DH, LQ, BATCH * HEADS, HEADS, 1.0f, 0);
        gemm(stream, CTX, 0, 0, EDIM, Wso, 0, 0, EDIM, 1, bso,
             TMP, 0, 0, EDIM, NTOK, EDIM, EDIM, 1, 1, 1.0f, 0);
        add_layernorm<<<NTOK, 256, 0, stream>>>(X, TMP, ln1w + l * EDIM, ln1b + l * EDIM, X);

        // ---- cross attention ----
        gemm(stream, X, 0, 0, EDIM, Wci, 0, 0, EDIM, 1, bci,
             QKV, 0, 0, EDIM, NTOK, EDIM, EDIM, 1, 1, 1.0f, 0);                 // Q
        gemm(stream, MEMB, 0, 0, EDIM, Wci + E2, 0, 0, EDIM, 1, bci + EDIM,
             KVK, 0, 0, EDIM, NTOK, EDIM, EDIM, 1, 1, 1.0f, 0);                 // K
        gemm(stream, MEMB, 0, 0, EDIM, Wci + 2 * E2, 0, 0, EDIM, 1, bci + 2 * EDIM,
             KVV, 0, 0, EDIM, NTOK, EDIM, EDIM, 1, 1, 1.0f, 0);                 // V
        gemm(stream, QKV, (long long)LQ * EDIM, DH, EDIM,
             KVK, (long long)LS * EDIM, DH, EDIM, 1, nullptr,
             SC, sBH, sHH, LQ, LQ, LS, DH, BATCH * HEADS, HEADS, 0.125f, 0);
        softmax_kernel<<<BATCH * HEADS * LQ, 128, 0, stream>>>(SC, nullptr, nullptr, mkpm);
        gemm(stream, SC, sBH, sHH, LQ,
             KVV, (long long)LS * EDIM, DH, 1, EDIM, nullptr,
             CTX, (long long)LQ * EDIM, DH, EDIM, LQ, DH, LS, BATCH * HEADS, HEADS, 1.0f, 0);
        gemm(stream, CTX, 0, 0, EDIM, Wco, 0, 0, EDIM, 1, bco,
             TMP, 0, 0, EDIM, NTOK, EDIM, EDIM, 1, 1, 1.0f, 0);
        add_layernorm<<<NTOK, 256, 0, stream>>>(X, TMP, ln2w + l * EDIM, ln2b + l * EDIM, X);

        // ---- feed-forward (exact gelu) ----
        gemm(stream, X, 0, 0, EDIM, l1w + (long long)l * FFDIM * EDIM, 0, 0, EDIM, 1,
             l1b + (long long)l * FFDIM,
             FF1, 0, 0, FFDIM, NTOK, FFDIM, EDIM, 1, 1, 1.0f, 1);
        gemm(stream, FF1, 0, 0, FFDIM, l2w + (long long)l * EDIM * FFDIM, 0, 0, FFDIM, 1,
             l2b + (long long)l * EDIM,
             TMP, 0, 0, EDIM, NTOK, EDIM, FFDIM, 1, 1, 1.0f, 0);
        add_layernorm<<<NTOK, 256, 0, stream>>>(X, TMP, ln3w + l * EDIM, ln3b + l * EDIM, X);
    }

    // ---- head ----
    add_layernorm<<<NTOK, 256, 0, stream>>>(X, nullptr, flnw, flnb, TMP);   // TMP = LN(x)
    // generator logits -> out[:, 0:80]
    gemm(stream, TMP, 0, 0, EDIM, genw, 0, 0, EDIM, 1, genb,
         out, 0, 0, OUTCOL, NTOK, V0C, EDIM, 1, 1, 1.0f, 0);
    // edge projections (reuse QKV region)
    float* E0 = QKV;
    float* E1 = QKV + 2 * MF;
    gemm(stream, TMP, 0, 0, EDIM, e0w, 0, 0, EDIM, 1, e0b,
         E0, 0, 0, EDIM, NTOK, EDIM, EDIM, 1, 1, 1.0f, 0);
    gemm(stream, TMP, 0, 0, EDIM, e1w, 0, 0, EDIM, 1, e1b,
         E1, 0, 0, EDIM, NTOK, EDIM, EDIM, 1, 1, 1.0f, 0);
    // edge logits -> out[:, 80:208], scale 1/sqrt(E) = 1/32
    gemm(stream, E0, (long long)LQ * EDIM, 0, EDIM,
         E1, (long long)LQ * EDIM, 0, EDIM, 1, nullptr,
         out + V0C, (long long)LQ * OUTCOL, 0, OUTCOL,
         LQ, LQ, EDIM, BATCH, 1, 1.0f / 32.0f, 0);
    // apply pred mask
    mask_kernel<<<(NTOK * OUTCOL) / 256, 256, 0, stream>>>(pmask, out);
}
